// MoE_32770600468772
// MI455X (gfx1250) — matrix operand order repaired and checked by reading, not compiled, not run on hardware
//
#include <hip/hip_runtime.h>
#include <hip/hip_bf16.h>

// ---------------------------------------------------------------------------
// MoE FFN for MI455X (gfx1250): top-2 routed experts + 2 shared experts.
// bf16 WMMA (v_wmma_f32_16x16x32_bf16) with f32 accumulation.
// Async global->LDS double-buffered pipeline (GLOBAL_LOAD_ASYNC_TO_LDS_B128).
// ---------------------------------------------------------------------------

#define TOKENS 2048   // B*S
#define DIM    1024   // D
#define FFN    4096   // F
#define NEXP   8      // routed experts
#define NALL   10     // routed + 2 shared

#define AS_GLOBAL __attribute__((address_space(1)))
#define AS_LDS    __attribute__((address_space(3)))

typedef int b128_t __attribute__((vector_size(16)));   // builtin's pointee type

#if defined(__HIP_DEVICE_COMPILE__) && \
    __has_builtin(__builtin_amdgcn_global_load_async_to_lds_b128)
#define USE_ASYNC 1
#else
#define USE_ASYNC 0
#endif

#if defined(__HIP_DEVICE_COMPILE__) && \
    __has_builtin(__builtin_amdgcn_s_wait_asynccnt)
#define WAIT_ASYNC(N) __builtin_amdgcn_s_wait_asynccnt(N)
#elif defined(__HIP_DEVICE_COMPILE__)
#define WAIT_ASYNC(N) asm volatile("s_wait_asynccnt %0" ::"i"(N) : "memory")
#else
#define WAIT_ASYNC(N)
#endif

typedef __attribute__((ext_vector_type(16))) __bf16 v16bf;
typedef __attribute__((ext_vector_type(8)))  __bf16 v8bf;
typedef __attribute__((ext_vector_type(8)))  float  v8f;

__device__ __forceinline__ __bf16 f2bf(float f) { return (__bf16)f; }

__device__ __forceinline__ float gelu_exact(float v) {
    return 0.5f * v * (1.0f + erff(v * 0.70710678118654752f));
}

// ---------------------------------------------------------------------------
// init: out = x (residual), xb = bf16(x), reset counts, shared-expert lists
// ---------------------------------------------------------------------------
__global__ __launch_bounds__(256) void moe_init(const float* __restrict__ x,
                                                float* __restrict__ out,
                                                __bf16* __restrict__ xb,
                                                int* __restrict__ cnt,
                                                int* __restrict__ perm,
                                                float* __restrict__ wgt) {
    unsigned i = blockIdx.x * 256u + threadIdx.x;      // grid covers TOKENS*DIM
    if (i < (unsigned)TOKENS * DIM) {
        float v = x[i];
        out[i] = v;                // residual init
        xb[i]  = f2bf(v);
    }
    if (i < (unsigned)NALL * TOKENS) {
        int e = i / TOKENS, t = i % TOKENS;
        if (e < NEXP) { perm[i] = 0; wgt[i] = 0.0f; }   // padding-safe defaults
        else          { perm[i] = t; wgt[i] = 1.0f; }   // shared experts: identity
    }
    if (i < 16) cnt[i] = (i == 8 || i == 9) ? TOKENS : 0;
}

// ---------------------------------------------------------------------------
// routing: one wave32 per token
// ---------------------------------------------------------------------------
__global__ __launch_bounds__(256) void moe_route(const float* __restrict__ x,
                                                 const float* __restrict__ cent,
                                                 const float* __restrict__ rbias,
                                                 int* __restrict__ cnt,
                                                 int* __restrict__ perm,
                                                 float* __restrict__ wgt) {
    int lane = threadIdx.x & 31;
    int wv   = threadIdx.x >> 5;
    int t    = blockIdx.x * 8 + wv;                    // grid = TOKENS/8 blocks

    float acc[NEXP];
    #pragma unroll
    for (int e = 0; e < NEXP; ++e) acc[e] = 0.0f;

    const float4* xv = (const float4*)(x + (size_t)t * DIM);
    for (int i = lane; i < DIM / 4; i += 32) {
        float4 xq = xv[i];
        #pragma unroll
        for (int e = 0; e < NEXP; ++e) {
            float4 cq = ((const float4*)(cent + (size_t)e * DIM))[i];
            acc[e] += xq.x * cq.x + xq.y * cq.y + xq.z * cq.z + xq.w * cq.w;
        }
    }
    #pragma unroll
    for (int e = 0; e < NEXP; ++e)
        #pragma unroll
        for (int off = 16; off > 0; off >>= 1)
            acc[e] += __shfl_xor(acc[e], off, 32);

    if (lane == 0) {
        float bal[NEXP];
        #pragma unroll
        for (int e = 0; e < NEXP; ++e) bal[e] = acc[e] + rbias[e];
        int i0 = 0;
        #pragma unroll
        for (int e = 1; e < NEXP; ++e) if (bal[e] > bal[i0]) i0 = e;
        int i1 = (i0 == 0) ? 1 : 0;
        #pragma unroll
        for (int e = 0; e < NEXP; ++e)
            if (e != i0 && bal[e] > bal[i1]) i1 = e;

        float s0 = 1.0f / (1.0f + expf(-acc[i0]));     // routing (unbiased) scores
        float s1 = 1.0f / (1.0f + expf(-acc[i1]));
        float mx = fmaxf(s0, s1);
        float e0 = expf(s0 - mx), e1 = expf(s1 - mx);
        float inv = 1.0f / (e0 + e1);

        int p0 = atomicAdd(&cnt[i0], 1);
        perm[i0 * TOKENS + p0] = t; wgt[i0 * TOKENS + p0] = e0 * inv;
        int p1 = atomicAdd(&cnt[i1], 1);
        perm[i1 * TOKENS + p1] = t; wgt[i1 * TOKENS + p1] = e1 * inv;
    }
}

// ---------------------------------------------------------------------------
// fragment loads + 4x2 WMMA for one 128x128x32 stage.
// A in LDS as bf16 (stride 40), B in LDS as raw f32 (stride 36), converted to
// bf16 at fragment build (co-executes with the XDL WMMAs).
// ---------------------------------------------------------------------------
__device__ __forceinline__ void compute_tile(const __bf16 (*__restrict__ As)[40],
                                             const float  (*__restrict__ Bs)[36],
                                             v8f (&acc)[4][2],
                                             int waveM, int waveN, int lr, int hl) {
    v16bf afrag[4];
    #pragma unroll
    for (int m = 0; m < 4; ++m) {
        const __bf16* p = &As[waveM * 64 + m * 16 + lr][hl * 8];
        v8bf lo = *(const v8bf*)p;
        v8bf hi = *(const v8bf*)(p + 16);
        afrag[m] = __builtin_shufflevector(lo, hi,
            0,1,2,3,4,5,6,7,8,9,10,11,12,13,14,15);
    }
    #pragma unroll
    for (int n = 0; n < 2; ++n) {
        const float* p = &Bs[waveN * 32 + n * 16 + lr][hl * 8];
        float4 f0 = *(const float4*)(p + 0);
        float4 f1 = *(const float4*)(p + 4);
        float4 f2 = *(const float4*)(p + 16);
        float4 f3 = *(const float4*)(p + 20);
        v16bf bfrag = { f2bf(f0.x), f2bf(f0.y), f2bf(f0.z), f2bf(f0.w),
                        f2bf(f1.x), f2bf(f1.y), f2bf(f1.z), f2bf(f1.w),
                        f2bf(f2.x), f2bf(f2.y), f2bf(f2.z), f2bf(f2.w),
                        f2bf(f3.x), f2bf(f3.y), f2bf(f3.z), f2bf(f3.w) };
        #pragma unroll
        for (int m = 0; m < 4; ++m)
            acc[m][n] = __builtin_amdgcn_wmma_f32_16x16x32_bf16(
                false, afrag[m], false, bfrag, (short)0, acc[m][n],
                false, false);
    }
}

// ---------------------------------------------------------------------------
// Tiled GEMM: out[m][n] = dot(A[m][:K], W[n][:K]) with bf16 WMMA, f32 acc.
// MODE 0 (fc):   A = xb gathered via perm, epilogue = GELU -> bf16 hbuf
// MODE 1 (proj): A = hbuf (expert-local rows), epilogue = *= wgt, atomic add out
// Block tile 128x128x32, 8 waves (2 M x 4 N), wave tile 64x32 = 4x2 WMMA frags.
// ---------------------------------------------------------------------------
template <int MODE>
__global__ __launch_bounds__(256) void moe_gemm(const __bf16* __restrict__ A,
                                                const float*  __restrict__ W,
                                                __bf16* __restrict__ hout,
                                                float* __restrict__ out,
                                                const int*   __restrict__ perm,
                                                const float* __restrict__ wgt,
                                                const int*   __restrict__ cnt,
                                                int Ktot, int lda, int Ntot) {
    const int ncnt = *cnt;
    const int m0 = blockIdx.y * 128;
    const int n0 = blockIdx.x * 128;
    if (m0 >= ncnt) return;                            // skip tiles past token count

    __shared__ __bf16 As[2][128][40];                  // padded: conflict-free b128 reads
    __shared__ float  Bs[2][128][36];

    const int tid  = threadIdx.x;
    const int lane = tid & 31;
    const int wv   = tid >> 5;
    const int waveM = wv >> 2;                         // 0..1
    const int waveN = wv & 3;                          // 0..3
    const int lr = lane & 15;
    const int hl = lane >> 4;                          // half-wave select

    v8f acc[4][2] = {};
    const int nsteps = Ktot / 32;

#if USE_ASYNC
    // -------- double-buffered async global->LDS pipeline --------
    // A slab: 128 rows x 64B = 512 x 16B chunks -> 2 per thread
    const int aRowLo = tid >> 2;                       // 0..63
    const int aSubB  = (tid & 3) * 16;                 // byte offset within 64B row
    const int tokLo  = (MODE == 0) ? perm[m0 + aRowLo]      : m0 + aRowLo;
    const int tokHi  = (MODE == 0) ? perm[m0 + 64 + aRowLo] : m0 + 64 + aRowLo;
    const char* aSrcLo = (const char*)(A + (size_t)tokLo * lda) + aSubB;
    const char* aSrcHi = (const char*)(A + (size_t)tokHi * lda) + aSubB;
    // B slab: 128 rows x 128B = 1024 x 16B chunks -> 4 per thread
    const int bRow  = tid >> 3;                        // 0..31 (+32 per chunk group)
    const int bSubB = (tid & 7) * 16;
    const char* bSrc0 = (const char*)(W + (size_t)(n0 + bRow)      * Ktot) + bSubB;
    const char* bSrc1 = (const char*)(W + (size_t)(n0 + bRow + 32) * Ktot) + bSubB;
    const char* bSrc2 = (const char*)(W + (size_t)(n0 + bRow + 64) * Ktot) + bSubB;
    const char* bSrc3 = (const char*)(W + (size_t)(n0 + bRow + 96) * Ktot) + bSubB;

    auto issue = [&](int buf, int k0) {
        const size_t aOff = (size_t)k0 * 2;            // bf16 bytes
        const size_t bOff = (size_t)k0 * 4;            // f32 bytes
        __builtin_amdgcn_global_load_async_to_lds_b128(
            (AS_GLOBAL b128_t*)(aSrcLo + aOff),
            (AS_LDS b128_t*)((char*)&As[buf][aRowLo][0] + aSubB), 0, 0);
        __builtin_amdgcn_global_load_async_to_lds_b128(
            (AS_GLOBAL b128_t*)(aSrcHi + aOff),
            (AS_LDS b128_t*)((char*)&As[buf][64 + aRowLo][0] + aSubB), 0, 0);
        __builtin_amdgcn_global_load_async_to_lds_b128(
            (AS_GLOBAL b128_t*)(bSrc0 + bOff),
            (AS_LDS b128_t*)((char*)&Bs[buf][bRow][0] + bSubB), 0, 0);
        __builtin_amdgcn_global_load_async_to_lds_b128(
            (AS_GLOBAL b128_t*)(bSrc1 + bOff),
            (AS_LDS b128_t*)((char*)&Bs[buf][bRow + 32][0] + bSubB), 0, 0);
        __builtin_amdgcn_global_load_async_to_lds_b128(
            (AS_GLOBAL b128_t*)(bSrc2 + bOff),
            (AS_LDS b128_t*)((char*)&Bs[buf][bRow + 64][0] + bSubB), 0, 0);
        __builtin_amdgcn_global_load_async_to_lds_b128(
            (AS_GLOBAL b128_t*)(bSrc3 + bOff),
            (AS_LDS b128_t*)((char*)&Bs[buf][bRow + 96][0] + bSubB), 0, 0);
    };

    issue(0, 0);
    for (int s = 0; s < nsteps; ++s) {
        const int buf = s & 1;
        if (s + 1 < nsteps) {
            issue(1 - buf, (s + 1) * 32);
            WAIT_ASYNC(6);                             // 6 newer in flight -> stage s done
        } else {
            WAIT_ASYNC(0);
        }
        __syncthreads();                               // publish stage s to all waves
        compute_tile(As[buf], Bs[buf], acc, waveM, waveN, lr, hl);
        __syncthreads();                               // all reads of buf done before reuse
    }
#else
    // -------- fallback: synchronous VGPR-staged loads, single buffer --------
    const int ar = tid >> 1;                           // 0..127
    const int ah = tid & 1;                            // 0/1
    const int grow = m0 + ar;
    const int tok  = (MODE == 0) ? perm[grow] : grow;
    const __bf16* aRow = A + (size_t)tok * lda;
    const float*  wRow = W + (size_t)(n0 + ar) * (size_t)Ktot;

    for (int k0 = 0; k0 < Ktot; k0 += 32) {
        {
            const uint4* g = (const uint4*)(aRow + k0 + ah * 16);
            uint4 q0 = g[0], q1 = g[1];
            *(uint4*)&As[0][ar][ah * 16]     = q0;
            *(uint4*)&As[0][ar][ah * 16 + 8] = q1;
        }
        {
            const float4* g = (const float4*)(wRow + k0 + ah * 16);
            float4 q0 = g[0], q1 = g[1], q2 = g[2], q3 = g[3];
            float* d = &Bs[0][ar][ah * 16];
            *(float4*)(d + 0)  = q0;
            *(float4*)(d + 4)  = q1;
            *(float4*)(d + 8)  = q2;
            *(float4*)(d + 12) = q3;
        }
        if (k0 + 32 < Ktot)
            __builtin_prefetch((const void*)(wRow + k0 + 32 + ah * 16), 0, 1);
        __syncthreads();
        compute_tile(As[0], Bs[0], acc, waveM, waveN, lr, hl);
        __syncthreads();
    }
#endif

    // ---- epilogue ----
    if (MODE == 0) {
        #pragma unroll
        for (int m = 0; m < 4; ++m) {
            int rb = m0 + waveM * 64 + m * 16 + hl * 8;
            #pragma unroll
            for (int n = 0; n < 2; ++n) {
                int cb = n0 + waveN * 32 + n * 16 + lr;
                #pragma unroll
                for (int r = 0; r < 8; ++r) {
                    float v = gelu_exact(acc[m][n][r]);
                    hout[(size_t)(rb + r) * Ntot + cb] = f2bf(v);
                }
            }
        }
    } else {
        #pragma unroll
        for (int m = 0; m < 4; ++m) {
            int rb = m0 + waveM * 64 + m * 16 + hl * 8;
            #pragma unroll
            for (int r = 0; r < 8; ++r) {
                int row = rb + r;
                int t   = perm[row];
                float w = wgt[row];
                if (w != 0.0f) {
                    #pragma unroll
                    for (int n = 0; n < 2; ++n) {
                        int cb = n0 + waveN * 32 + n * 16 + lr;
                        unsafeAtomicAdd(out + (size_t)t * Ntot + cb,
                                        acc[m][n][r] * w);
                    }
                }
            }
        }
    }
}

// ---------------------------------------------------------------------------
// host launcher
// ---------------------------------------------------------------------------
extern "C" void kernel_launch(void* const* d_in, const int* in_sizes, int n_in,
                              void* d_out, int out_size, void* d_ws, size_t ws_size,
                              hipStream_t stream) {
    const float* x       = (const float*)d_in[0];
    const float* cent    = (const float*)d_in[1];
    const float* rbias   = (const float*)d_in[2];
    const float* Wfc_r   = (const float*)d_in[3];
    const float* Wproj_r = (const float*)d_in[4];
    const float* Wfc_sh  = (const float*)d_in[5];
    const float* Wproj_sh= (const float*)d_in[6];
    float* out = (float*)d_out;

    char* ws = (char*)d_ws;
    __bf16* xb = (__bf16*)ws;                                    //  4 MB
    __bf16* hb = (__bf16*)(ws + ((size_t)4 << 20));              // 16 MB
    char* meta = ws + ((size_t)20 << 20);
    int*   perm = (int*)meta;                                    // NALL*TOKENS ints
    float* wgt  = (float*)(meta + (size_t)NALL * TOKENS * 4);
    int*   cnt  = (int*)(meta + (size_t)2 * NALL * TOKENS * 4);

    moe_init<<<(TOKENS * DIM) / 256, 256, 0, stream>>>(x, out, xb, cnt, perm, wgt);
    moe_route<<<TOKENS / 8, 256, 0, stream>>>(x, cent, rbias, cnt, perm, wgt);

    for (int e = 0; e < NALL; ++e) {
        const float* wfc = (e < NEXP)
            ? Wfc_r   + (size_t)e * FFN * DIM
            : Wfc_sh  + (size_t)(e - NEXP) * FFN * DIM;
        const float* wpr = (e < NEXP)
            ? Wproj_r + (size_t)e * DIM * FFN
            : Wproj_sh+ (size_t)(e - NEXP) * DIM * FFN;
        const int*   pe = perm + e * TOKENS;
        const float* we = wgt  + e * TOKENS;
        const int*   ce = cnt  + e;

        dim3 gf(FFN / 128, TOKENS / 128);   // 32 x 16
        moe_gemm<0><<<gf, 256, 0, stream>>>(xb, wfc, hb, nullptr, pe, we, ce,
                                            DIM, DIM, FFN);
        dim3 gp(DIM / 128, TOKENS / 128);   // 8 x 16
        moe_gemm<1><<<gp, 256, 0, stream>>>(hb, wpr, nullptr, out, pe, we, ce,
                                            FFN, FFN, DIM);
    }
}
